// MultiheadLSHSelfAttention_78907139162461
// MI455X (gfx1250) — matrix operand order repaired and checked by reading, not compiled, not run on hardware
//
#include <hip/hip_runtime.h>
#include <math.h>

#define NN 4
#define TT 4096
#define DD 512
#define HH 8
#define NHASH 4
#define BSZ 64
#define DHD 64
#define NBK 64
#define CSZ 256      // NHASH*NBK chunks
#define LLN 16384    // NHASH*TT sorted length per head

typedef __attribute__((ext_vector_type(16))) __bf16 v16bf;
typedef __attribute__((ext_vector_type(8)))  __bf16 v8bf;
typedef __attribute__((ext_vector_type(4)))  __bf16 v4bf;
typedef __attribute__((ext_vector_type(8)))  float  v8f;
typedef __attribute__((ext_vector_type(4)))  unsigned int v4u;
typedef __attribute__((ext_vector_type(8)))  int v8i;
typedef __attribute__((ext_vector_type(4)))  int v4i;

union V16  { v16bf v; v8bf h[2]; v4bf q[4]; __bf16 e[16]; };
union V8   { v8f  v; float  e[8];  };
union V4B  { v4bf v; __bf16 e[4];  };

#if __has_builtin(__builtin_amdgcn_tensor_load_to_lds) && __has_builtin(__builtin_amdgcn_s_wait_tensorcnt)
#define USE_TDM 1
#else
#define USE_TDM 0
#endif

// ---------------- WMMA fragment loaders (ISA lane layouts) ----------------
// A (16x32 bf16): lane row = lane&15; two contiguous 8-elem runs at
// k = 8*half and k = 16+8*half.  p points at row_base + ks*32.
__device__ __forceinline__ v16bf fragA16(const __bf16* p, int hv) {   // 16B-aligned rows
    V16 r;
    r.h[0] = *(const v8bf*)(p + 8 * hv);
    r.h[1] = *(const v8bf*)(p + 16 + 8 * hv);
    return r.v;
}
// B (32x16 bf16): lane col = lane&15; two contiguous 8-elem runs at
// k = 8*half and k = 16+8*half (same order as A).
__device__ __forceinline__ v16bf fragB16(const __bf16* p, int hv) {   // 16B-aligned rows
    V16 r;
    r.h[0] = *(const v8bf*)(p + 8 * hv);
    r.h[1] = *(const v8bf*)(p + 16 + 8 * hv);
    return r.v;
}
__device__ __forceinline__ v16bf fragA8(const __bf16* p, int hv) {    // 8B-aligned rows
    V16 r;
    r.q[0] = *(const v4bf*)(p + 8 * hv);
    r.q[1] = *(const v4bf*)(p + 8 * hv + 4);
    r.q[2] = *(const v4bf*)(p + 16 + 8 * hv);
    r.q[3] = *(const v4bf*)(p + 16 + 8 * hv + 4);
    return r.v;
}
__device__ __forceinline__ v16bf fragB8(const __bf16* p, int hv) {    // 8B-aligned rows
    V16 r;
    r.q[0] = *(const v4bf*)(p + 8 * hv);
    r.q[1] = *(const v4bf*)(p + 8 * hv + 4);
    r.q[2] = *(const v4bf*)(p + 16 + 8 * hv);
    r.q[3] = *(const v4bf*)(p + 16 + 8 * hv + 4);
    return r.v;
}

// ---------------- wave32 half-group (16-lane) reductions ----------------
__device__ __forceinline__ float hred_max(float v) {
    v = fmaxf(v, __shfl_xor(v, 1, 32));
    v = fmaxf(v, __shfl_xor(v, 2, 32));
    v = fmaxf(v, __shfl_xor(v, 4, 32));
    v = fmaxf(v, __shfl_xor(v, 8, 32));
    return v;
}
__device__ __forceinline__ float hred_sum(float v) {
    v += __shfl_xor(v, 1, 32);
    v += __shfl_xor(v, 2, 32);
    v += __shfl_xor(v, 4, 32);
    v += __shfl_xor(v, 8, 32);
    return v;
}

#if USE_TDM
// Issue a 2D TDM tile load (bf16, data_size=2B) with LDS padding of 4 dwords
// every 32 dwords stored -> LDS rows land at 72-element (144B) stride.
__device__ __forceinline__ void tdm_load_2d(unsigned ldsOff, const void* gptr,
                                            unsigned tensor_d0, unsigned tensor_d1,
                                            unsigned tile_d0, unsigned tile_d1,
                                            unsigned stride0)
{
    unsigned long long ga = (unsigned long long)gptr;
    v4u g0;
    g0[0] = 1u;                                   // count=1, user descriptor
    g0[1] = ldsOff;                               // LDS byte address
    g0[2] = (unsigned)ga;                         // global_addr[31:0]
    g0[3] = ((unsigned)(ga >> 32) & 0x1FFFFFFu) | (2u << 30);  // addr[56:32] | type=2
    v8i g1;
    // data_size=1 (2B) @[17:16], pad_enable @20, pad_interval=4 (32 dwords) @[24:22],
    // pad_amount=3 (4 dwords) @[31:25]
    g1[0] = (int)((1u << 16) | (1u << 20) | (4u << 22) | (3u << 25));
    g1[1] = (int)((tensor_d0 & 0xFFFFu) << 16);                     // dim0[15:0] @ bit48
    g1[2] = (int)((tensor_d0 >> 16) | ((tensor_d1 & 0xFFFFu) << 16));
    g1[3] = (int)((tensor_d1 >> 16) | (tile_d0 << 16));
    g1[4] = (int)tile_d1;                                            // tile_dim2 = 0
    g1[5] = (int)stride0;                                            // dim0 stride low32
    g1[6] = 0;
    g1[7] = 0;                                                       // dim1 stride = 0 (2D)
    v4i gz4 = {0, 0, 0, 0};
    v8i gz8 = {0, 0, 0, 0, 0, 0, 0, 0};
    // 6-arg form (this toolchain): (g0, g1, g2, g3, extra_group, cpol)
    __builtin_amdgcn_tensor_load_to_lds(g0, g1, gz4, gz4, gz8, 0);
}
#endif

// =====================================================================
// Kernel 0: one-time precision prep:
//   xb  = bf16(x)                 [16384][512]
//   wqt = bf16(W_Q^T), wvt = bf16(W_V^T)   [n][k]
// =====================================================================
__global__ __launch_bounds__(256)
void k_prep(const float* __restrict__ x,
            const float* __restrict__ Wq, const float* __restrict__ Wv,
            __bf16* __restrict__ xb, __bf16* __restrict__ wqt, __bf16* __restrict__ wvt)
{
    long i = (long)blockIdx.x * 256 + threadIdx.x;
    const long XN = (long)NN * TT * DD;
    const long WN = (long)DD * DD;
    if (i < XN) {
        xb[i] = (__bf16)x[i];
    } else if (i < XN + WN) {
        long o = i - XN;
        int n = (int)(o >> 9), k = (int)(o & 511);
        wqt[o] = (__bf16)Wq[(size_t)k * DD + n];
    } else if (i < XN + 2 * WN) {
        long o = i - XN - WN;
        int n = (int)(o >> 9), k = (int)(o & 511);
        wvt[o] = (__bf16)Wv[(size_t)k * DD + n];
    }
}

// =====================================================================
// Kernel 1: projection GEMM  out = xb @ W^T(bf16,[n][k]) + bias.
// 128x64 block tile, K-step 64, tiles staged in LDS via TDM (or
// vectorized fallback), fragments read as b128/b64 LDS vectors.
// =====================================================================
__global__ __launch_bounds__(256)
void k_proj(const __bf16* __restrict__ xb, const __bf16* __restrict__ wt,
            const float* __restrict__ bias, float* __restrict__ out)
{
    __shared__ __align__(16) __bf16 ta[128][72];   // A tile (rows of x)
    __shared__ __align__(16) __bf16 tb[64][72];    // B tile (rows = output col n)

    int tid  = threadIdx.x;
    int wave = tid >> 5, lane = tid & 31;
    int hv = lane >> 4, ln = lane & 15;
    int nb = blockIdx.x;     // 0..7   : 64-wide col block
    int mb = blockIdx.y;     // 0..127 : 128-tall row block

    v8f acc[4] = {{}, {}, {}, {}};

    for (int kb = 0; kb < DD; kb += 64) {
        __syncthreads();                       // previous tiles fully consumed
#if USE_TDM
        if (tid == 0) {
            tdm_load_2d((unsigned)(size_t)&ta[0][0],
                        xb + (size_t)(mb * 128) * DD + kb,
                        DD, (unsigned)(NN * TT), 64, 128, DD);
            tdm_load_2d((unsigned)(size_t)&tb[0][0],
                        wt + (size_t)(nb * 64) * DD + kb,
                        DD, DD, 64, 64, DD);
        }
        if (tid < 32) __builtin_amdgcn_s_wait_tensorcnt(0);
#else
        for (int i = tid; i < 128 * 8; i += 256) {
            int r = i >> 3, cj = i & 7;
            *(v8bf*)&ta[r][cj * 8] =
                *(const v8bf*)(xb + (size_t)(mb * 128 + r) * DD + kb + cj * 8);
        }
        for (int i = tid; i < 64 * 8; i += 256) {
            int r = i >> 3, cj = i & 7;
            *(v8bf*)&tb[r][cj * 8] =
                *(const v8bf*)(wt + (size_t)(nb * 64 + r) * DD + kb + cj * 8);
        }
#endif
        __syncthreads();
        #pragma unroll
        for (int ks = 0; ks < 2; ++ks) {
            v16bf a = fragA16(&ta[wave * 16 + ln][ks * 32], hv);
            #pragma unroll
            for (int nt = 0; nt < 4; ++nt) {
                v16bf b = fragB16(&tb[nt * 16 + ln][ks * 32], hv);
                acc[nt] = __builtin_amdgcn_wmma_f32_16x16x32_bf16(false, a, false, b,
                                                                  (short)0, acc[nt], false, false);
            }
        }
    }
    #pragma unroll
    for (int nt = 0; nt < 4; ++nt) {
        V8 c; c.v = acc[nt];
        int ncol = nb * 64 + nt * 16 + ln;
        float bnv = bias[ncol];
        #pragma unroll
        for (int r = 0; r < 8; ++r) {
            int m = mb * 128 + wave * 16 + r + 8 * hv;
            out[(size_t)m * DD + ncol] = c.e[r] + bnv;
        }
    }
}

// =====================================================================
// Kernel 2: LSH hashing, one thread per (head,b,hash,t).
// =====================================================================
__global__ void k_hash(const float* __restrict__ Q, const float* __restrict__ rot,
                       int* __restrict__ buckets)
{
    long g = (long)blockIdx.x * blockDim.x + threadIdx.x;
    int t    = (int)(g % TT);
    int nh   = (int)((g / TT) % NHASH);
    int b    = (int)((g / ((long)TT * NHASH)) % NN);
    int head = (int)(g / ((long)TT * NHASH * NN));

    float r[32];
    #pragma unroll
    for (int i = 0; i < 32; ++i) r[i] = 0.f;

    const float4* qp4 = (const float4*)(Q + ((size_t)(b * TT + t)) * DD + head * DHD);
    for (int f4 = 0; f4 < 16; ++f4) {
        float4 q = qp4[f4];
        const float* rp = rot + ((size_t)((head * DHD + f4 * 4) * NHASH + nh)) * 32;
        float qv[4] = {q.x, q.y, q.z, q.w};
        #pragma unroll
        for (int u = 0; u < 4; ++u) {
            const float* rpu = rp + (size_t)u * NHASH * 32;
            #pragma unroll
            for (int i = 0; i < 32; ++i) r[i] = fmaf(qv[u], rpu[i], r[i]);
        }
    }
    float best = r[0]; int bi = 0;
    #pragma unroll
    for (int i = 1; i < 32; ++i) if (r[i]  > best) { best = r[i];  bi = i;      }
    #pragma unroll
    for (int i = 0; i < 32; ++i) if (-r[i] > best) { best = -r[i]; bi = 32 + i; }

    buckets[((size_t)(head * NN + b)) * LLN + nh * TT + t] = bi + nh * NBK;
}

// =====================================================================
// Kernel 3: stable counting sort per (head,batch).
// =====================================================================
__global__ __launch_bounds__(256)
void k_sort(const int* __restrict__ buckets,
            int* __restrict__ sticker, int* __restrict__ sbuck,
            int* __restrict__ undo)
{
    __shared__ unsigned char lb[LLN];
    __shared__ int cnt[256];

    size_t base = (size_t)blockIdx.x * LLN;   // blockIdx.x = head*NN + b
    int tid = threadIdx.x;

    for (int i = tid; i < LLN; i += 256) lb[i] = (unsigned char)buckets[base + i];
    __syncthreads();

    int my = 0;
    for (int l = 0; l < LLN; ++l) if ((int)lb[l] == tid) my++;
    cnt[tid] = my;
    __syncthreads();
    for (int off = 1; off < 256; off <<= 1) {
        int v = (tid >= off) ? cnt[tid - off] : 0;
        __syncthreads();
        cnt[tid] += v;
        __syncthreads();
    }
    int pos = cnt[tid] - my;                  // exclusive prefix
    for (int l = 0; l < LLN; ++l) {
        if ((int)lb[l] == tid) {
            sticker[base + pos] = l;
            sbuck[base + pos]   = tid;
            undo[base + l]      = pos;
            pos++;
        }
    }
}

// =====================================================================
// Kernel 4: fused chunk attention (one 256-thread block per chunk).
// =====================================================================
__global__ __launch_bounds__(256)
void k_attn(const float* __restrict__ Q, const float* __restrict__ V,
            const int* __restrict__ sticker, const int* __restrict__ sbuck,
            float* __restrict__ so, float* __restrict__ slog)
{
    __shared__ __align__(16) __bf16 bqs[64][72];    // queries
    __shared__ __align__(16) __bf16 bks[128][72];   // normalized keys (cur|prev)
    __shared__ __align__(16) __bf16 bvT[64][132];   // V transposed: [n][k]
    __shared__ __align__(16) __bf16 pbs[64][132];   // probs (bf16)
    __shared__ float  rowmax[64][2], rowsum[64][2];
    __shared__ float  gmaxA[64], grcpA[64];
    __shared__ int    rb[64], rposA[64], cbk[128], cposA[128];

    int bx   = blockIdx.x;
    int c    = bx & (CSZ - 1);
    int hb   = bx >> 8;                  // head*NN + b
    int head = hb / NN, b = hb % NN;
    size_t sbase = (size_t)hb * LLN;
    int tid = threadIdx.x;

    // ---- stage ----
    if (tid < 64) {
        int p  = sticker[sbase + (size_t)c * BSZ + tid];
        int t0 = p % TT;
        rposA[tid] = t0;
        rb[tid]    = sbuck[sbase + (size_t)c * BSZ + tid];
        const float4* qp4 = (const float4*)(Q + ((size_t)(b * TT + t0)) * DD + head * DHD);
        #pragma unroll
        for (int f4 = 0; f4 < 16; ++f4) {
            float4 q = qp4[f4];
            V4B w;
            w.e[0] = (__bf16)q.x; w.e[1] = (__bf16)q.y;
            w.e[2] = (__bf16)q.z; w.e[3] = (__bf16)q.w;
            *(v4bf*)&bqs[tid][f4 * 4] = w.v;
        }
    } else if (tid < 192) {
        int j   = tid - 64;                              // 0..127
        int cc  = (j < 64) ? c : ((c + CSZ - 1) & (CSZ - 1));
        int off = j & 63;
        int p   = sticker[sbase + (size_t)cc * BSZ + off];
        int t0  = p % TT;
        cposA[j] = t0;
        cbk[j]   = sbuck[sbase + (size_t)cc * BSZ + off];
        const float4* qp4 = (const float4*)(Q + ((size_t)(b * TT + t0)) * DD + head * DHD);
        const float4* vp4 = (const float4*)(V + ((size_t)(b * TT + t0)) * DD + head * DHD);
        float nrm = 0.f;
        #pragma unroll
        for (int f4 = 0; f4 < 16; ++f4) {
            float4 q = qp4[f4];
            nrm = fmaf(q.x, q.x, fmaf(q.y, q.y, fmaf(q.z, q.z, fmaf(q.w, q.w, nrm))));
        }
        float inv = 1.f / (sqrtf(nrm) + 1e-6f);
        #pragma unroll
        for (int f4 = 0; f4 < 16; ++f4) {
            float4 q = qp4[f4];
            V4B w;
            w.e[0] = (__bf16)(q.x * inv); w.e[1] = (__bf16)(q.y * inv);
            w.e[2] = (__bf16)(q.z * inv); w.e[3] = (__bf16)(q.w * inv);
            *(v4bf*)&bks[j][f4 * 4] = w.v;
            float4 vv = vp4[f4];
            bvT[f4 * 4 + 0][j] = (__bf16)vv.x;
            bvT[f4 * 4 + 1][j] = (__bf16)vv.y;
            bvT[f4 * 4 + 2][j] = (__bf16)vv.z;
            bvT[f4 * 4 + 3][j] = (__bf16)vv.w;
        }
    }
    __syncthreads();

    int wave = tid >> 5, lane = tid & 31;
    int hv = lane >> 4, ln = lane & 15;
    int mt = wave >> 1;                  // M tile
    int nhf = wave & 1;                  // N half
    const float scale = 0.125f;          // DH^-0.5

    // ---- dots = bq @ bk^T (registers) ----
    V8 dot[4];
    #pragma unroll
    for (int ntl = 0; ntl < 4; ++ntl) {
        int nt = nhf * 4 + ntl;
        v8f acc = {};
        #pragma unroll
        for (int ks = 0; ks < 2; ++ks) {
            v16bf a = fragA16(&bqs[mt * 16 + ln][ks * 32], hv);
            v16bf bmat = fragB16(&bks[nt * 16 + ln][ks * 32], hv);
            acc = __builtin_amdgcn_wmma_f32_16x16x32_bf16(false, a, false, bmat,
                                                          (short)0, acc, false, false);
        }
        V8 cc_; cc_.v = acc;
        int n = nt * 16 + ln;
        #pragma unroll
        for (int r = 0; r < 8; ++r) {
            int m = mt * 16 + r + 8 * hv;
            float v = cc_.e[r] * scale;
            if (rb[m] != cbk[n])      v = -1e9f;
            if (rposA[m] == cposA[n]) v = -1e-5f;
            dot[ntl].e[r] = v;
        }
    }

    // ---- softmax partials via shuffles ----
    float pm[8], ps[8];
    #pragma unroll
    for (int r = 0; r < 8; ++r) {
        float m4 = fmaxf(fmaxf(dot[0].e[r], dot[1].e[r]), fmaxf(dot[2].e[r], dot[3].e[r]));
        pm[r] = hred_max(m4);
        float e4 = __expf(dot[0].e[r] - pm[r]) + __expf(dot[1].e[r] - pm[r])
                 + __expf(dot[2].e[r] - pm[r]) + __expf(dot[3].e[r] - pm[r]);
        ps[r] = hred_sum(e4);
    }
    #pragma unroll
    for (int r = 0; r < 8; ++r) {
        if (ln == r) {
            int m = mt * 16 + r + 8 * hv;
            rowmax[m][nhf] = pm[r];
            rowsum[m][nhf] = ps[r];
        }
    }
    __syncthreads();

    if (tid < 64) {
        float m0 = rowmax[tid][0], m1 = rowmax[tid][1];
        float gm = fmaxf(m0, m1);
        float gs = rowsum[tid][0] * __expf(m0 - gm) + rowsum[tid][1] * __expf(m1 - gm);
        gmaxA[tid] = gm;
        grcpA[tid] = 1.f / gs;
        slog[sbase + (size_t)c * BSZ + tid] = gm + __logf(gs);
    }
    __syncthreads();

    #pragma unroll
    for (int ntl = 0; ntl < 4; ++ntl) {
        int n = (nhf * 4 + ntl) * 16 + ln;
        #pragma unroll
        for (int r = 0; r < 8; ++r) {
            int m = mt * 16 + r + 8 * hv;
            pbs[m][n] = (__bf16)(__expf(dot[ntl].e[r] - gmaxA[m]) * grcpA[m]);
        }
    }
    __syncthreads();

    // ---- out = probs @ bv ----
    int ntb = (wave & 1) * 2;
    #pragma unroll
    for (int q2 = 0; q2 < 2; ++q2) {
        int nt = ntb + q2;
        v8f acc = {};
        #pragma unroll
        for (int ks = 0; ks < 4; ++ks) {
            v16bf a = fragA8(&pbs[mt * 16 + ln][ks * 32], hv);
            v16bf bmat = fragB8(&bvT[nt * 16 + ln][ks * 32], hv);
            acc = __builtin_amdgcn_wmma_f32_16x16x32_bf16(false, a, false, bmat,
                                                          (short)0, acc, false, false);
        }
        V8 cc_; cc_.v = acc;
        #pragma unroll
        for (int r = 0; r < 8; ++r) {
            int m = mt * 16 + r + 8 * hv;
            so[(sbase + (size_t)c * BSZ + m) * DHD + nt * 16 + ln] = cc_.e[r];
        }
    }
}

// =====================================================================
// Kernel 5: unsort + hash-softmax combine + LayerNorm.
// =====================================================================
__global__ __launch_bounds__(256)
void k_combine(const float* __restrict__ so, const float* __restrict__ slog,
               const int* __restrict__ undo,
               const float* __restrict__ gamma, const float* __restrict__ beta,
               float* __restrict__ out)
{
    __shared__ float red[256];
    int bt = blockIdx.x;
    int b = bt / TT, t = bt % TT;
    int tid = threadIdx.x;

    float acc2[2];
    float s1 = 0.f, s2 = 0.f;
    #pragma unroll
    for (int j = 0; j < 2; ++j) {
        int d = tid + 256 * j;
        int head = d >> 6, dh = d & 63;
        size_t hb = (size_t)(head * NN + b) * LLN;
        float lg[NHASH]; int pp[NHASH];
        float mx = -1e30f;
        #pragma unroll
        for (int h = 0; h < NHASH; ++h) {
            int p = undo[hb + (size_t)h * TT + t];
            pp[h] = p;
            lg[h] = slog[hb + p];
            mx = fmaxf(mx, lg[h]);
        }
        float ssum = 0.f;
        #pragma unroll
        for (int h = 0; h < NHASH; ++h) { lg[h] = __expf(lg[h] - mx); ssum += lg[h]; }
        float inv = 1.f / ssum, a = 0.f;
        #pragma unroll
        for (int h = 0; h < NHASH; ++h)
            a += so[(hb + (size_t)pp[h]) * DHD + dh] * lg[h] * inv;
        acc2[j] = a;
        s1 += a; s2 += a * a;
    }

    red[tid] = s1; __syncthreads();
    for (int s = 128; s > 0; s >>= 1) { if (tid < s) red[tid] += red[tid + s]; __syncthreads(); }
    float mu = red[0] * (1.f / 512.f);
    __syncthreads();
    red[tid] = s2; __syncthreads();
    for (int s = 128; s > 0; s >>= 1) { if (tid < s) red[tid] += red[tid + s]; __syncthreads(); }
    float var = red[0] * (1.f / 512.f) - mu * mu;
    float rstd = rsqrtf(var + 1e-3f);

    #pragma unroll
    for (int j = 0; j < 2; ++j) {
        int d = tid + 256 * j;
        out[(size_t)bt * DD + d] = gamma[d] * (acc2[j] - mu) * rstd + beta[d];
    }
}

// =====================================================================
extern "C" void kernel_launch(void* const* d_in, const int* in_sizes, int n_in,
                              void* d_out, int out_size, void* d_ws, size_t ws_size,
                              hipStream_t stream)
{
    (void)in_sizes; (void)n_in; (void)out_size; (void)ws_size;
    const float* x     = (const float*)d_in[0];
    const float* Wq    = (const float*)d_in[1];
    const float* bq    = (const float*)d_in[2];
    const float* Wv    = (const float*)d_in[3];
    const float* bv    = (const float*)d_in[4];
    const float* gamma = (const float*)d_in[5];
    const float* beta  = (const float*)d_in[6];
    const float* rot   = (const float*)d_in[7];

    char* ws = (char*)d_ws;
    size_t off = 0;
    float*  Qb      = (float*) (ws + off); off += (size_t)NN * TT * DD * 4;
    float*  Vb      = (float*) (ws + off); off += (size_t)NN * TT * DD * 4;
    __bf16* xbf     = (__bf16*)(ws + off); off += (size_t)NN * TT * DD * 2;
    __bf16* wqt     = (__bf16*)(ws + off); off += (size_t)DD * DD * 2;
    __bf16* wvt     = (__bf16*)(ws + off); off += (size_t)DD * DD * 2;
    int*    buckets = (int*)   (ws + off); off += (size_t)HH * NN * LLN * 4;
    int*    sticker = (int*)   (ws + off); off += (size_t)HH * NN * LLN * 4;
    int*    sbuck   = (int*)   (ws + off); off += (size_t)HH * NN * LLN * 4;
    int*    undo    = (int*)   (ws + off); off += (size_t)HH * NN * LLN * 4;
    float*  so      = (float*) (ws + off); off += (size_t)HH * NN * LLN * DHD * 4;
    float*  slog    = (float*) (ws + off); off += (size_t)HH * NN * LLN * 4;

    const long XN = (long)NN * TT * DD, WN = (long)DD * DD;
    int prep_blocks = (int)((XN + 2 * WN + 255) / 256);
    k_prep<<<prep_blocks, 256, 0, stream>>>(x, Wq, Wv, xbf, wqt, wvt);

    dim3 gp(8, 128);
    k_proj<<<gp, 256, 0, stream>>>(xbf, wqt, bq, Qb);
    k_proj<<<gp, 256, 0, stream>>>(xbf, wvt, bv, Vb);

    k_hash<<<2048, 256, 0, stream>>>(Qb, rot, buckets);
    k_sort<<<HH * NN, 256, 0, stream>>>(buckets, sticker, sbuck, undo);
    k_attn<<<HH * NN * CSZ, 256, 0, stream>>>(Qb, Vb, sticker, sbuck, so, slog);
    k_combine<<<NN * TT, 256, 0, stream>>>(so, slog, undo, gamma, beta, (float*)d_out);
}
